// Decoder_34325378630277
// MI455X (gfx1250) — compile-verified, matrix operand order checked by reading, not run on hardware
//
#include <hip/hip_runtime.h>
#include <hip/hip_bf16.h>
#include <math.h>

typedef __bf16  v16bf __attribute__((ext_vector_type(16)));
typedef float   v8f   __attribute__((ext_vector_type(8)));

#define TWO_PI 6.28318530717958647692f

static __device__ __forceinline__ unsigned short f2bf(float f) {
  unsigned int u = __float_as_uint(f);
  return (unsigned short)((u + 0x7FFFu + ((u >> 16) & 1u)) >> 16);  // RNE
}
static __device__ __forceinline__ float bf2f(unsigned short b) {
  return __uint_as_float(((unsigned int)b) << 16);
}

union FragBF { v16bf v; uint4 q[2]; };

// Load one 16x32 bf16 fragment (per-lane: two contiguous 16B chunks).
// base already includes the c0 chunk offset; khalf = lane>>4.
template<int C>
static __device__ __forceinline__ void load_frag(FragBF& f,
    const unsigned short* __restrict__ base, int khalf)
{
  const int k0 = khalf * 8;
  f.q[0] = *reinterpret_cast<const uint4*>(base + k0);
  if constexpr (C > 16)
    f.q[1] = *reinterpret_cast<const uint4*>(base + 16 + k0);
  else
    f.q[1] = make_uint4(0u, 0u, 0u, 0u);
}

static __device__ __forceinline__ v8f wmma_bf16(const FragBF& a, const FragBF& b, v8f c) {
  return __builtin_amdgcn_wmma_f32_16x16x32_bf16(
             false, a.v, false, b.v, (short)0, c, false, false);
}

// ---------------------------------------------------------------------------
// NCHW f32 -> padded NHWC bf16 (B, H+2, W+2, C), interior only
// ---------------------------------------------------------------------------
__global__ void __launch_bounds__(256) nchw_to_nhwc_bf16(
    const float* __restrict__ in, unsigned short* __restrict__ out,
    int Bn, int C, int H, int W)
{
  __shared__ unsigned short tile[32][33];
  const int nWT = W >> 5;
  const int nCT = (C + 31) >> 5;
  int t = blockIdx.x;
  const int wt = t % nWT; t /= nWT;
  const int ct = t % nCT; t /= nCT;
  const int h  = t % H;   t /= H;
  const int b  = t;
  const int tx = threadIdx.x & 31;
  const int ty = threadIdx.x >> 5;
  const int P = W + 2;
  for (int i = ty; i < 32; i += 8) {
    const int c = ct*32 + i;
    const int w = wt*32 + tx;
    tile[i][tx] = (c < C) ? f2bf(in[(((size_t)b*C + c)*H + h)*W + w]) : (unsigned short)0;
  }
  __syncthreads();
  for (int i = ty; i < 32; i += 8) {
    const int w = wt*32 + i;
    const int c = ct*32 + tx;
    if (c < C)
      out[(((size_t)b*(H+2) + (h+1))*P + (w+1))*C + c] = tile[tx][i];
  }
}

// Zero the 1-pixel halo of a padded NHWC buffer (layout depends on C!)
__global__ void __launch_bounds__(256) zero_border(
    unsigned short* __restrict__ buf, int Bn, int H, int W, int C)
{
  const int P = W + 2;
  const int npix = 2*P + 2*H;
  int idx = blockIdx.x * blockDim.x + threadIdx.x;
  int total = Bn * npix * C;
  if (idx >= total) return;
  int c = idx % C; int t = idx / C;
  int p = t % npix; int b = t / npix;
  int h, w;
  if (p < P)            { h = 0;     w = p; }
  else if (p < 2*P)     { h = H + 1; w = p - P; }
  else { int q = p - 2*P; h = 1 + (q >> 1); w = (q & 1) ? (W + 1) : 0; }
  buf[((size_t)(b*(H+2) + h)*P + w)*C + c] = 0;
}

// ---------------------------------------------------------------------------
// Weight prep
// ---------------------------------------------------------------------------
__global__ void __launch_bounds__(256) prep_conv3_weights(
    const float* __restrict__ w, unsigned short* __restrict__ out,
    int S, int OC, int IC)
{
  int idx = blockIdx.x * blockDim.x + threadIdx.x;
  int total = S * 9 * OC * IC;
  if (idx >= total) return;
  int ic = idx % IC; int t = idx / IC;
  int oc = t % OC;   t /= OC;
  int p  = t % 9;    int s = t / 9;
  out[(((size_t)s*9 + p)*OC + oc)*IC + ic] =
      f2bf(w[(((size_t)s*OC + oc)*IC + ic)*9 + p]);
}

__global__ void __launch_bounds__(256) prep_conv1_weights(
    const float* __restrict__ w, unsigned short* __restrict__ out, int OC, int IC)
{
  int idx = blockIdx.x * blockDim.x + threadIdx.x;
  if (idx >= OC * IC) return;
  out[idx] = f2bf(w[idx]);
}

// twiddles e^{+2pi i kx w / W} for kx=1..3 : tw[(kx-1)*W + w] = (cos, sin)
__global__ void __launch_bounds__(256) prep_twiddle(float2* __restrict__ tw, int W)
{
  int idx = blockIdx.x * blockDim.x + threadIdx.x;
  if (idx >= 3 * W) return;
  int kx = idx / W + 1;
  int w  = idx % W;
  float s, c;
  __sincosf(TWO_PI * (float)kx * (float)w / (float)W, &s, &c);
  tw[idx] = make_float2(c, s);
}

// ---------------------------------------------------------------------------
// Truncated forward DFT along W (padded NHWC input)
// ---------------------------------------------------------------------------
__global__ void __launch_bounds__(256) dft_w_kernel(
    const unsigned short* __restrict__ x, float2* __restrict__ xh,
    int Bn, int H, int W, int C)
{
  int idx = blockIdx.x * blockDim.x + threadIdx.x;
  int total = Bn * H * 4 * C;
  if (idx >= total) return;
  int c  = idx % C; int t = idx / C;
  int kx = t % 4;   t /= 4;
  int h  = t % H;   int b = t / H;
  const unsigned short* row =
      x + ((size_t)(b*(H+2) + h + 1)*(W+2) + 1)*C + c;
  float sr = 0.f, si = 0.f;
  const float step = -TWO_PI * (float)kx / (float)W;
  for (int w = 0; w < W; ++w) {
    float xv = bf2f(row[(size_t)w * C]);
    float s, cc; __sincosf(step * (float)w, &s, &cc);
    sr += xv * cc; si += xv * s;
  }
  xh[((size_t)(b*C + c)*H + h)*4 + kx] = make_float2(sr, si);
}

__global__ void __launch_bounds__(256) dft_h_kernel(
    const float2* __restrict__ xh, float2* __restrict__ xf,
    int Bn, int H, int C)
{
  int idx = blockIdx.x * blockDim.x + threadIdx.x;
  int total = Bn * C * 16;
  if (idx >= total) return;
  int kk = idx % 16; int t = idx / 16;
  int c  = t % C;    int b = t / C;
  int ky = kk >> 2, kx = kk & 3;
  const float2* col = xh + ((size_t)(b*C + c)*H)*4 + kx;
  float ar = 0.f, ai = 0.f;
  const float step = -TWO_PI * (float)ky / (float)H;
  for (int hh = 0; hh < H; ++hh) {
    float s, cc; __sincosf(step * (float)hh, &s, &cc);
    float2 v = col[(size_t)hh * 4];
    ar += v.x * cc - v.y * s;
    ai += v.x * s  + v.y * cc;
  }
  xf[((size_t)(b*C + c))*16 + kk] = make_float2(ar, ai);
}

__global__ void __launch_bounds__(256) spectral_mm_kernel(
    const float2* __restrict__ xf, const float* __restrict__ wr,
    const float* __restrict__ wi, float2* __restrict__ of,
    int Bn, int C, int OC)
{
  int idx = blockIdx.x * blockDim.x + threadIdx.x;
  int total = Bn * OC * 16;
  if (idx >= total) return;
  int kk = idx % 16; int t = idx / 16;
  int o  = t % OC;   int b = t / OC;
  float ar = 0.f, ai = 0.f;
  for (int i = 0; i < C; ++i) {
    float2 xv = xf[((size_t)(b*C + i))*16 + kk];
    float wrv = wr[((size_t)i*OC + o)*16 + kk];
    float wiv = wi[((size_t)i*OC + o)*16 + kk];
    ar += xv.x * wrv - xv.y * wiv;
    ai += xv.x * wiv + xv.y * wrv;
  }
  of[((size_t)(b*OC + o))*16 + kk] = make_float2(ar, ai);
}

// folds the (1/sqrt(HW))^2 ortho norm into g
__global__ void __launch_bounds__(256) idft_h_kernel(
    const float2* __restrict__ of, float2* __restrict__ g,
    int Bn, int H, int OC)
{
  int idx = blockIdx.x * blockDim.x + threadIdx.x;
  int total = Bn * OC * H * 4;
  if (idx >= total) return;
  int kx = idx & 3; int t = idx >> 2;
  int hh = t % H;   t /= H;
  int o  = t % OC;  int b = t / OC;
  float ar = 0.f, ai = 0.f;
  #pragma unroll
  for (int ky = 0; ky < 4; ++ky) {
    float2 v = of[((size_t)(b*OC + o))*16 + ky*4 + kx];
    float s, cc; __sincosf(TWO_PI * (float)ky * (float)hh / (float)H, &s, &cc);
    ar += v.x * cc - v.y * s;
    ai += v.x * s  + v.y * cc;
  }
  const float nrm = 1.0f / 65536.0f;
  g[(((size_t)(b*OC + o))*H + hh)*4 + kx] = make_float2(ar * nrm, ai * nrm);
}

// ---------------------------------------------------------------------------
// 3x3 conv + bias + ReLU over zero-padded NHWC bf16.
// One wave: 32 px x OCT oc (OCT=32 when C>=32, else 16).
// 4 WMMA per 8 b128 loads in the wide case.
// ---------------------------------------------------------------------------
template<int C>
__global__ void __launch_bounds__(256) conv3x3_wmma(
    const unsigned short* __restrict__ x,   // (B,H+2,W+2,C) bf16
    const unsigned short* __restrict__ wT,  // [9][C][C] bf16
    const float* __restrict__ bias,
    unsigned short* __restrict__ out,       // (B,H+2,W+2,C) bf16
    int Bn, int H, int W)
{
  constexpr int OCT = (C >= 32) ? 32 : 16;
  const int lane  = threadIdx.x & 31;
  const int wave  = blockIdx.x * 8 + (threadIdx.x >> 5);
  const int nOcg  = C / OCT;
  const int nWt   = W >> 5;
  int t = wave;
  const int ocg = t % nOcg; t /= nOcg;
  const int wt  = t % nWt;  t /= nWt;
  const int h   = t % H;    t /= H;
  const int b   = t;

  const int m     = lane & 15;
  const int khalf = lane >> 4;
  const int oc0   = ocg * OCT;
  const int P = W + 2;

  v8f acc00 = {}, acc10 = {}, acc01 = {}, acc11 = {};

  #pragma unroll
  for (int dy = 0; dy < 3; ++dy) {
    #pragma unroll
    for (int dx = 0; dx < 3; ++dx) {
      const unsigned short* xp =
          x + ((size_t)(b*(H+2) + h + dy)*P + (wt*32 + m + dx)) * C;
      const unsigned short* wp0 =
          wT + ((size_t)(dy*3 + dx)*C + (oc0 + m)) * C;
      #pragma unroll
      for (int c0 = 0; c0 < C; c0 += 32) {
        FragBF fa0, fa1, fb0;
        load_frag<C>(fa0, xp + c0, khalf);
        load_frag<C>(fa1, xp + (size_t)16*C + c0, khalf);
        load_frag<C>(fb0, wp0 + c0, khalf);
        acc00 = wmma_bf16(fa0, fb0, acc00);
        acc10 = wmma_bf16(fa1, fb0, acc10);
        if constexpr (OCT == 32) {
          FragBF fb1;
          load_frag<C>(fb1, wp0 + (size_t)16*C + c0, khalf);
          acc01 = wmma_bf16(fa0, fb1, acc01);
          acc11 = wmma_bf16(fa1, fb1, acc11);
        }
      }
    }
  }

  const float bv0 = bias[oc0 + m];
  unsigned short* op =
      out + ((size_t)(b*(H+2) + h + 1)*P + (wt*32 + 1)) * C + oc0 + m;
  #pragma unroll
  for (int r = 0; r < 8; ++r) {
    const int mm = r + khalf*8;
    op[(size_t)mm * C]        = f2bf(fmaxf(acc00[r] + bv0, 0.0f));
    op[(size_t)(mm + 16) * C] = f2bf(fmaxf(acc10[r] + bv0, 0.0f));
  }
  if constexpr (OCT == 32) {
    const float bv1 = bias[oc0 + 16 + m];
    #pragma unroll
    for (int r = 0; r < 8; ++r) {
      const int mm = r + khalf*8;
      op[(size_t)mm * C + 16]        = f2bf(fmaxf(acc01[r] + bv1, 0.0f));
      op[(size_t)(mm + 16) * C + 16] = f2bf(fmaxf(acc11[r] + bv1, 0.0f));
    }
  }
}

// ---------------------------------------------------------------------------
// 1x1 conv (WMMA) + fused FNO epilogue: out = gelu(conv+bias+idft_w(g)) + skip
// One wave: 32 px x OCT oc. Twiddles from a small table.
// ---------------------------------------------------------------------------
template<int C, int OC>
__global__ void __launch_bounds__(256) conv1x1_fno_wmma(
    const unsigned short* __restrict__ x, const unsigned short* __restrict__ wB,
    const float* __restrict__ bias, const float2* __restrict__ g,
    const float2* __restrict__ tw, const unsigned short* __restrict__ skip,
    unsigned short* __restrict__ out, int Bn, int H, int W)
{
  constexpr int OCT = (OC >= 32) ? 32 : 16;
  const int lane  = threadIdx.x & 31;
  const int wave  = blockIdx.x * 8 + (threadIdx.x >> 5);
  const int nOcg  = OC / OCT;
  const int nWt   = W >> 5;
  int t = wave;
  const int ocg = t % nOcg; t /= nOcg;
  const int wt  = t % nWt;  t /= nWt;
  const int h   = t % H;    t /= H;
  const int b   = t;

  const int m     = lane & 15;
  const int khalf = lane >> 4;
  const int oc0   = ocg * OCT;
  const int P = W + 2;

  v8f acc00 = {}, acc10 = {}, acc01 = {}, acc11 = {};
  const unsigned short* xp =
      x + ((size_t)(b*(H+2) + h + 1)*P + (wt*32 + m + 1)) * C;
  const unsigned short* wp0 = wB + (size_t)(oc0 + m) * C;
  #pragma unroll
  for (int c0 = 0; c0 < C; c0 += 32) {
    FragBF fa0, fa1, fb0;
    load_frag<C>(fa0, xp + c0, khalf);
    load_frag<C>(fa1, xp + (size_t)16*C + c0, khalf);
    load_frag<C>(fb0, wp0 + c0, khalf);
    acc00 = wmma_bf16(fa0, fb0, acc00);
    acc10 = wmma_bf16(fa1, fb0, acc10);
    if constexpr (OCT == 32) {
      FragBF fb1;
      load_frag<C>(fb1, wp0 + (size_t)16*C + c0, khalf);
      acc01 = wmma_bf16(fa0, fb1, acc01);
      acc11 = wmma_bf16(fa1, fb1, acc11);
    }
  }

  const size_t obase = ((size_t)(b*(H+2) + h + 1)*P + (wt*32 + 1)) * OC + oc0 + m;
  unsigned short*       op = out  + obase;
  const unsigned short* sp = skip + obase;

  #pragma unroll
  for (int nh = 0; nh < (OCT == 32 ? 2 : 1); ++nh) {
    const int oc = oc0 + m + nh*16;
    const float bv = bias[oc];
    const float2* gp = g + ((size_t)(b*OC + oc)*H + h) * 4;
    const float2 g0 = gp[0], g1 = gp[1], g2 = gp[2], g3 = gp[3];
    #pragma unroll
    for (int r = 0; r < 8; ++r) {
      #pragma unroll
      for (int half = 0; half < 2; ++half) {
        const int mm   = r + khalf*8 + half*16;
        const int wpix = wt*32 + mm;
        const float2 t1 = tw[wpix];
        const float2 t2 = tw[W + wpix];
        const float2 t3 = tw[2*W + wpix];
        float spec = g0.x
                   + 2.f*(g1.x*t1.x - g1.y*t1.y)
                   + 2.f*(g2.x*t2.x - g2.y*t2.y)
                   + 2.f*(g3.x*t3.x - g3.y*t3.y);
        float a;
        if (nh == 0) a = half ? acc10[r] : acc00[r];
        else         a = half ? acc11[r] : acc01[r];
        float v  = a + bv + spec;
        float ge = 0.5f * v * (1.0f + erff(v * 0.70710678118f));
        op[(size_t)mm * OC + nh*16] = f2bf(ge + bf2f(sp[(size_t)mm * OC + nh*16]));
      }
    }
  }
}

// ---------------------------------------------------------------------------
// Final 1x1 conv 16->2, f32 NCHW output
// ---------------------------------------------------------------------------
__global__ void __launch_bounds__(256) final_conv1x1(
    const unsigned short* __restrict__ x, const float* __restrict__ ow,
    const float* __restrict__ ob, float* __restrict__ out,
    int Bn, int H, int W)
{
  int idx = blockIdx.x * blockDim.x + threadIdx.x;
  int total = Bn * 2 * H * W;
  if (idx >= total) return;
  int w = idx % W; int t = idx / W;
  int h = t % H;   t /= H;
  int oc = t % 2;  int b = t / 2;
  const unsigned short* px =
      x + ((size_t)(b*(H+2) + h + 1)*(W+2) + (w + 1)) * 16;
  float s = ob[oc];
  #pragma unroll
  for (int i = 0; i < 16; ++i) s += bf2f(px[i]) * ow[oc*16 + i];
  out[((size_t)(b*2 + oc)*H + h)*W + w] = s;
}

// ---------------------------------------------------------------------------
// host-side helpers
// ---------------------------------------------------------------------------
static inline int cdiv_h(int a, int b) { return (a + b - 1) / b; }

template<int C, int OC>
static void run_fno(const unsigned short* xin, const float* wr, const float* wi,
                    const float* bw, const float* bb,
                    const unsigned short* skip, unsigned short* outp,
                    float2* xh, float2* xf, float2* ofb, float2* gb,
                    unsigned short* w1, const float2* tw,
                    int B, int H, int W, hipStream_t stream)
{
  constexpr int OCT = (OC >= 32) ? 32 : 16;
  { int tot = B*H*4*C;  dft_w_kernel<<<cdiv_h(tot,256),256,0,stream>>>(xin, xh, B,H,W,C); }
  { int tot = B*C*16;   dft_h_kernel<<<cdiv_h(tot,256),256,0,stream>>>(xh, xf, B,H,C); }
  { int tot = B*OC*16;  spectral_mm_kernel<<<cdiv_h(tot,256),256,0,stream>>>(xf, wr, wi, ofb, B,C,OC); }
  { int tot = B*OC*H*4; idft_h_kernel<<<cdiv_h(tot,256),256,0,stream>>>(ofb, gb, B,H,OC); }
  { int tot = OC*C;     prep_conv1_weights<<<cdiv_h(tot,256),256,0,stream>>>(bw, w1, OC, C); }
  { int waves = B*H*(W/32)*(OC/OCT);
    conv1x1_fno_wmma<C,OC><<<waves/8,256,0,stream>>>(xin, w1, bb, gb, tw, skip, outp, B,H,W); }
}

template<int C>
static unsigned short* run_cblock(const float* wAll, const float* bAll, int S,
                                  unsigned short* src, unsigned short* aux,
                                  unsigned short* w3,
                                  int B, int H, int W, hipStream_t stream)
{
  constexpr int OCT = (C >= 32) ? 32 : 16;
  { int npix = 2*(W+2) + 2*H; int tot = B*npix*C;
    zero_border<<<cdiv_h(tot,256),256,0,stream>>>(src, B,H,W,C);
    zero_border<<<cdiv_h(tot,256),256,0,stream>>>(aux, B,H,W,C); }
  { int tot = S*9*C*C;
    prep_conv3_weights<<<cdiv_h(tot,256),256,0,stream>>>(wAll, w3, S, C, C); }
  unsigned short* a = src; unsigned short* d = aux;
  for (int s = 0; s < S; ++s) {
    int waves = B*H*(W/32)*(C/OCT);
    conv3x3_wmma<C><<<waves/8,256,0,stream>>>(a, w3 + (size_t)s*9*C*C,
                                              bAll + s*C, d, B,H,W);
    unsigned short* tm = a; a = d; d = tm;
  }
  return a;
}

// ---------------------------------------------------------------------------
extern "C" void kernel_launch(void* const* d_in, const int* in_sizes, int n_in,
                              void* d_out, int out_size, void* d_ws, size_t ws_size,
                              hipStream_t stream)
{
  (void)in_sizes; (void)n_in; (void)out_size; (void)ws_size;
  const int B = 2, H = 256, W = 256;
  const size_t PIX = (size_t)(H + 2) * (W + 2);

  const float* x1 = (const float*)d_in[0];
  const float* x2 = (const float*)d_in[1];
  const float* x3 = (const float*)d_in[2];
  const float* x4 = (const float*)d_in[3];
  const float* x5 = (const float*)d_in[4];
  const float* f5_wr = (const float*)d_in[5],  *f5_wi = (const float*)d_in[6];
  const float* f5_bw = (const float*)d_in[7],  *f5_bb = (const float*)d_in[8];
  const float* f6_wr = (const float*)d_in[9],  *f6_wi = (const float*)d_in[10];
  const float* f6_bw = (const float*)d_in[11], *f6_bb = (const float*)d_in[12];
  const float* f7_wr = (const float*)d_in[13], *f7_wi = (const float*)d_in[14];
  const float* f7_bw = (const float*)d_in[15], *f7_bb = (const float*)d_in[16];
  const float* f8_wr = (const float*)d_in[17], *f8_wi = (const float*)d_in[18];
  const float* f8_bw = (const float*)d_in[19], *f8_bb = (const float*)d_in[20];
  const float* c6_w  = (const float*)d_in[21], *c6_b  = (const float*)d_in[22];
  const float* c7_w  = (const float*)d_in[23], *c7_b  = (const float*)d_in[24];
  const float* c8_w  = (const float*)d_in[25], *c8_b  = (const float*)d_in[26];
  const float* c9_w  = (const float*)d_in[27], *c9_b  = (const float*)d_in[28];
  const float* owp   = (const float*)d_in[29], *obp   = (const float*)d_in[30];

  char* p = (char*)d_ws;
  auto take = [&](size_t bytes) -> char* {
    char* r = p; p += (bytes + 255) & ~(size_t)255; return r;
  };
  unsigned short* x5s  = (unsigned short*)take((size_t)B*PIX*256*2);
  unsigned short* x4s  = (unsigned short*)take((size_t)B*PIX*128*2);
  unsigned short* x3s  = (unsigned short*)take((size_t)B*PIX*64*2);
  unsigned short* x2s  = (unsigned short*)take((size_t)B*PIX*32*2);
  unsigned short* x1s  = (unsigned short*)take((size_t)B*PIX*16*2);
  unsigned short* actA = (unsigned short*)take((size_t)B*PIX*128*2);
  unsigned short* actB = (unsigned short*)take((size_t)B*PIX*128*2);
  float2* xh  = (float2*)take((size_t)B*256*H*4*sizeof(float2));
  float2* xf  = (float2*)take((size_t)B*256*16*sizeof(float2));
  float2* ofb = (float2*)take((size_t)B*128*16*sizeof(float2));
  float2* gb  = (float2*)take((size_t)B*128*H*4*sizeof(float2));
  unsigned short* w3 = (unsigned short*)take((size_t)3*9*128*128*2);
  unsigned short* w1 = (unsigned short*)take((size_t)256*128*2);
  float2* tw = (float2*)take((size_t)3*W*sizeof(float2));

  // prep: layout conversions + twiddles
  auto cvt = [&](const float* in, unsigned short* out, int C) {
    int blocks = (W/32) * cdiv_h(C, 32) * B * H;
    nchw_to_nhwc_bf16<<<blocks, 256, 0, stream>>>(in, out, B, C, H, W);
  };
  cvt(x5, x5s, 256); cvt(x4, x4s, 128); cvt(x3, x3s, 64);
  cvt(x2, x2s, 32);  cvt(x1, x1s, 16);
  prep_twiddle<<<cdiv_h(3*W,256),256,0,stream>>>(tw, W);

  // pipeline
  run_fno<256,128>(x5s, f5_wr, f5_wi, f5_bw, f5_bb, x4s, actA,
                   xh, xf, ofb, gb, w1, tw, B,H,W, stream);          // x5u -> actA
  unsigned short* x6 = run_cblock<128>(c6_w, c6_b, 3, actA, actB, w3, B,H,W, stream);
  unsigned short* o1 = (x6 == actA) ? actB : actA;
  run_fno<128,64>(x6, f6_wr, f6_wi, f6_bw, f6_bb, x3s, o1,
                  xh, xf, ofb, gb, w1, tw, B,H,W, stream);           // x6u -> o1
  unsigned short* x7 = run_cblock<64>(c7_w, c7_b, 3, o1, x6, w3, B,H,W, stream);
  unsigned short* o2 = (x7 == actA) ? actB : actA;
  run_fno<64,32>(x7, f7_wr, f7_wi, f7_bw, f7_bb, x2s, o2,
                 xh, xf, ofb, gb, w1, tw, B,H,W, stream);            // x7u -> o2
  unsigned short* x8 = run_cblock<32>(c8_w, c8_b, 2, o2, x7, w3, B,H,W, stream);
  unsigned short* o3 = (x8 == actA) ? actB : actA;
  run_fno<32,16>(x8, f8_wr, f8_wi, f8_bw, f8_bb, x1s, o3,
                 xh, xf, ofb, gb, w1, tw, B,H,W, stream);            // x8u -> o3
  unsigned short* x9 = run_cblock<16>(c9_w, c9_b, 1, o3, x8, w3, B,H,W, stream);

  { int tot = B*2*H*W;
    final_conv1x1<<<cdiv_h(tot,256),256,0,stream>>>(x9, owp, obp, (float*)d_out, B,H,W); }
}
